// OptLayer_3307124818391
// MI455X (gfx1250) — matrix operand order repaired and checked by reading, not compiled, not run on hardware
//
#include <hip/hip_runtime.h>

// ---- CDNA5 WMMA vector types (wave32) ----
typedef __attribute__((ext_vector_type(16))) __bf16 v16bf;
typedef __attribute__((ext_vector_type(8)))  __bf16 v8bf;
typedef __attribute__((ext_vector_type(8)))  float  v8f;

#define B_ROWS 65536
#define K_DIM  512
#define N_DIM  256
#define M_TILE 128
#define NTHREADS 512
#define KT     32
#define LDW    40   // ushort elems per LDS row: 32 data + 8 pad = 80 bytes (16B aligned, 20-bank stride)
#define W_ELEMS (N_DIM * K_DIM)

#define S_CONST  0.1f
#define R2_CONST 0.02f
#define EPS_F    1e-12f

__device__ __forceinline__ unsigned short f2bf_rne(float f) {
    unsigned u = __float_as_uint(f);
    u += 0x7FFFu + ((u >> 16) & 1u);          // round-to-nearest-even
    return (unsigned short)(u >> 16);
}
__device__ __forceinline__ float bf2f(unsigned short h) {
    return __uint_as_float(((unsigned)h) << 16);
}

// split float4 -> bf16 hi plane + bf16 lo plane (error-compensated), 8B LDS stores
__device__ __forceinline__ void stage4(float4 v, unsigned short* hi, unsigned short* lo) {
    unsigned short h0 = f2bf_rne(v.x), h1 = f2bf_rne(v.y);
    unsigned short h2 = f2bf_rne(v.z), h3 = f2bf_rne(v.w);
    float r0 = v.x - bf2f(h0), r1 = v.y - bf2f(h1);
    float r2 = v.z - bf2f(h2), r3 = v.w - bf2f(h3);
    unsigned short l0 = f2bf_rne(r0), l1 = f2bf_rne(r1);
    unsigned short l2 = f2bf_rne(r2), l3 = f2bf_rne(r3);
    uint2 hp, lp;
    hp.x = (unsigned)h0 | ((unsigned)h1 << 16); hp.y = (unsigned)h2 | ((unsigned)h3 << 16);
    lp.x = (unsigned)l0 | ((unsigned)l1 << 16); lp.y = (unsigned)l2 | ((unsigned)l3 << 16);
    *(uint2*)hi = hp;
    *(uint2*)lo = lp;
}

#define CAT16(a,b) __builtin_shufflevector(a, b, 0,1,2,3,4,5,6,7,8,9,10,11,12,13,14,15)

// One-time W pre-split: f32 -> bf16 hi plane + bf16 lo plane in workspace.
__global__ __launch_bounds__(256)
void split_w_kernel(const float* __restrict__ W, unsigned short* __restrict__ wpl) {
    int idx = blockIdx.x * 256 + threadIdx.x;
    if (idx < W_ELEMS) {
        float f = W[idx];
        unsigned short h = f2bf_rne(f);
        float r = f - bf2f(h);
        wpl[idx]           = h;
        wpl[W_ELEMS + idx] = f2bf_rne(r);
    }
}

// PRE=true : W planes pre-split in global; DMA them global->LDS asynchronously.
// PRE=false: convert W in-kernel (fallback if workspace too small).
template <bool PRE>
__global__ __launch_bounds__(NTHREADS)
void proj_gemm_wmma(const float* __restrict__ x,
                    const float* __restrict__ W,
                    const float* __restrict__ b,
                    const unsigned short* __restrict__ wpl,
                    float* __restrict__ out)
{
    __shared__ __attribute__((aligned(16))) unsigned short xs_hi[M_TILE * LDW];
    __shared__ __attribute__((aligned(16))) unsigned short xs_lo[M_TILE * LDW];
    __shared__ __attribute__((aligned(16))) unsigned short ws_hi[N_DIM * LDW];
    __shared__ __attribute__((aligned(16))) unsigned short ws_lo[N_DIM * LDW];
    __shared__ float sum_t[M_TILE];
    __shared__ float sum_zz[M_TILE];

    const int tid   = threadIdx.x;
    const int lane  = tid & 31;
    const int wave  = tid >> 5;          // 0..15
    const int waveM = wave >> 2;         // 0..3 : which 32-row slice of the 128-row tile
    const int waveN = wave & 3;          // 0..3 : which 64-col slice of N
    const int half  = lane >> 4;         // 0/1  : lane half (WMMA layout)
    const int l16   = lane & 15;

    const int rowBase = blockIdx.x * M_TILE;

    if (tid < M_TILE) { sum_t[tid] = 0.f; sum_zz[tid] = 0.f; }

    v8f acc[2][4];
    const v8f z8 = {0.f,0.f,0.f,0.f,0.f,0.f,0.f,0.f};
    #pragma unroll
    for (int mt = 0; mt < 2; ++mt)
        #pragma unroll
        for (int nt = 0; nt < 4; ++nt) acc[mt][nt] = z8;

    for (int kk = 0; kk < K_DIM / KT; ++kk) {
        const int k0 = kk * KT;
        __syncthreads();                       // previous tile fully consumed

        // ---- stage W block: 256 rows x 32 K, hi+lo planes ----
        if (PRE) {
            // async DMA of pre-split planes straight into padded LDS layout
            // 256 rows x 4 x 16B chunks per plane = 1024 chunks; 512 threads -> 2 each
            #pragma unroll
            for (int i = 0; i < 2; ++i) {
                int chunk = tid + i * NTHREADS;      // 0..1023
                int r = chunk >> 2, c = chunk & 3;
                unsigned lds_hi = (unsigned)(uintptr_t)(ws_hi + r * LDW + c * 8);
                unsigned lds_lo = (unsigned)(uintptr_t)(ws_lo + r * LDW + c * 8);
                const unsigned short* ghi = wpl + (size_t)r * K_DIM + k0 + c * 8;
                const unsigned short* glo = ghi + W_ELEMS;
                asm volatile("global_load_async_to_lds_b128 %0, %1, off"
                             :: "v"(lds_hi), "v"(ghi) : "memory");
                asm volatile("global_load_async_to_lds_b128 %0, %1, off"
                             :: "v"(lds_lo), "v"(glo) : "memory");
            }
        } else {
            #pragma unroll
            for (int i = 0; i < 4; ++i) {
                int chunk = tid + i * NTHREADS;      // 0..2047 float4 chunks
                int r = chunk >> 3, c = chunk & 7;
                float4 v = ((const float4*)(W + (size_t)r * K_DIM + k0))[c];
                stage4(v, ws_hi + r * LDW + c * 4, ws_lo + r * LDW + c * 4);
            }
        }

        // ---- stage x block: 128 rows x 32 K (convert f32 -> bf16 hi/lo) ----
        #pragma unroll
        for (int i = 0; i < 2; ++i) {
            int chunk = tid + i * NTHREADS;          // 0..1023
            int r = chunk >> 3, c = chunk & 7;
            float4 v = ((const float4*)(x + (size_t)(rowBase + r) * K_DIM + k0))[c];
            stage4(v, xs_hi + r * LDW + c * 4, xs_lo + r * LDW + c * 4);
        }
        // prefetch next x K-tile (global_prefetch_b8)
        if (kk + 1 < K_DIM / KT) {
            int r = tid >> 3, c = tid & 7;
            __builtin_prefetch(x + (size_t)(rowBase + r) * K_DIM + k0 + KT + c * 4, 0, 0);
        }
        if (PRE) asm volatile("s_wait_asynccnt 0" ::: "memory");
        __syncthreads();

        // ---- WMMA: acc += A_hi*B_hi + A_hi*B_lo + A_lo*B_hi ----
        #pragma unroll
        for (int mt = 0; mt < 2; ++mt) {
            const int ar = ((waveM * 2 + mt) * 16 + l16) * LDW + half * 16;
            v8bf ahl = *(const v8bf*)(xs_hi + ar);
            v8bf ahh = *(const v8bf*)(xs_hi + ar + 8);
            v8bf all = *(const v8bf*)(xs_lo + ar);
            v8bf alh = *(const v8bf*)(xs_lo + ar + 8);
            v16bf a_hi = CAT16(ahl, ahh);
            v16bf a_lo = CAT16(all, alh);
            #pragma unroll
            for (int nt = 0; nt < 4; ++nt) {
                const int wr = ((waveN * 4 + nt) * 16 + l16) * LDW + half * 16;
                v8bf whl = *(const v8bf*)(ws_hi + wr);
                v8bf whh = *(const v8bf*)(ws_hi + wr + 8);
                v8bf wll = *(const v8bf*)(ws_lo + wr);
                v8bf wlh = *(const v8bf*)(ws_lo + wr + 8);
                v16bf w_hi = CAT16(whl, whh);
                v16bf w_lo = CAT16(wll, wlh);
                acc[mt][nt] = __builtin_amdgcn_wmma_f32_16x16x32_bf16(
                    false, a_hi, false, w_hi, (short)0, acc[mt][nt], false, false);
                acc[mt][nt] = __builtin_amdgcn_wmma_f32_16x16x32_bf16(
                    false, a_hi, false, w_lo, (short)0, acc[mt][nt], false, false);
                acc[mt][nt] = __builtin_amdgcn_wmma_f32_16x16x32_bf16(
                    false, a_lo, false, w_hi, (short)0, acc[mt][nt], false, false);
            }
        }
    }

    // ---- bias: z = x@W^T - b ----
    float bn[4];
    #pragma unroll
    for (int nt = 0; nt < 4; ++nt) bn[nt] = b[(waveN * 4 + nt) * 16 + l16];
    #pragma unroll
    for (int mt = 0; mt < 2; ++mt)
        #pragma unroll
        for (int nt = 0; nt < 4; ++nt)
            #pragma unroll
            for (int r = 0; r < 8; ++r) acc[mt][nt][r] -= bn[nt];

    // ---- row stats t = sum(z), zz = sum(z^2) ----
    // C layout: VGPR r, lanes 0-15 -> row r, lanes 16-31 -> row r+8; col = l16.
    #pragma unroll
    for (int mt = 0; mt < 2; ++mt) {
        #pragma unroll
        for (int r = 0; r < 8; ++r) {
            float v = 0.f, vv = 0.f;
            #pragma unroll
            for (int nt = 0; nt < 4; ++nt) {
                float z = acc[mt][nt][r];
                v += z; vv += z * z;
            }
            #pragma unroll
            for (int off = 1; off < 16; off <<= 1) {   // stays inside 16-lane halves
                v  += __shfl_xor(v,  off, 32);
                vv += __shfl_xor(vv, off, 32);
            }
            if (l16 == 0) {
                int Mrow = (waveM * 2 + mt) * 16 + half * 8 + r;
                atomicAdd(&sum_t[Mrow],  v);   // ds_add_f32
                atomicAdd(&sum_zz[Mrow], vv);
            }
        }
    }
    __syncthreads();

    // ---- projection epilogue: y = alpha*z + beta (3-case KKT) ----
    const float nF = (float)N_DIM;
    #pragma unroll
    for (int mt = 0; mt < 2; ++mt) {
        #pragma unroll
        for (int r = 0; r < 8; ++r) {
            const int Mrow = (waveM * 2 + mt) * 16 + half * 8 + r;
            const float t  = sum_t[Mrow];
            const float zz = sum_zz[Mrow];
            // case 1: slab projection
            float tc    = fminf(fmaxf(t, -S_CONST), S_CONST);
            float beta1 = (tc - t) / nF;
            float y1sq  = zz + 2.f * beta1 * t + nF * beta1 * beta1;
            bool  ok1   = y1sq <= R2_CONST;
            // case 2: ball projection
            float znorm = sqrtf(fmaxf(zz, EPS_F));
            float scale = fminf(1.f, sqrtf(R2_CONST) / znorm);
            bool  ok2   = fabsf(t) * scale <= S_CONST;
            // case 3: both active
            float denom = fmaxf(nF * zz - t * t, EPS_F);
            float c3    = sqrtf(fmaxf(nF * R2_CONST - S_CONST * S_CONST, 0.f) / denom);
            float sp    = (t > 0.f) ? S_CONST : ((t < 0.f) ? -S_CONST : 0.f);
            float beta3 = (sp - c3 * t) / nF;

            float alpha = ok1 ? 1.f   : (ok2 ? scale : c3);
            float beta  = ok1 ? beta1 : (ok2 ? 0.f   : beta3);

            const size_t orow = (size_t)(rowBase + Mrow) * N_DIM;
            #pragma unroll
            for (int nt = 0; nt < 4; ++nt) {
                int col = (waveN * 4 + nt) * 16 + l16;
                out[orow + col] = alpha * acc[mt][nt][r] + beta;
            }
        }
    }
}

extern "C" void kernel_launch(void* const* d_in, const int* in_sizes, int n_in,
                              void* d_out, int out_size, void* d_ws, size_t ws_size,
                              hipStream_t stream) {
    (void)in_sizes; (void)n_in; (void)out_size;
    const float* x = (const float*)d_in[0];
    const float* W = (const float*)d_in[1];
    const float* b = (const float*)d_in[2];
    float* out = (float*)d_out;

    const size_t need = (size_t)2 * W_ELEMS * sizeof(unsigned short);   // 512 KB
    dim3 grid(B_ROWS / M_TILE);                                        // 512 blocks x 512 threads

    if (ws_size >= need) {
        unsigned short* wpl = (unsigned short*)d_ws;
        split_w_kernel<<<dim3((W_ELEMS + 255) / 256), dim3(256), 0, stream>>>(W, wpl);
        proj_gemm_wmma<true><<<grid, dim3(NTHREADS), 0, stream>>>(x, W, b, wpl, out);
    } else {
        proj_gemm_wmma<false><<<grid, dim3(NTHREADS), 0, stream>>>(x, W, b, nullptr, out);
    }
}
